// AGL_2869038154058
// MI455X (gfx1250) — matrix operand order repaired and checked by reading, not compiled, not run on hardware
//
#include <hip/hip_runtime.h>

// ---------------------------------------------------------------------------
// AV-localization transformer forward on MI455X (gfx1250, wave32, WMMA bf16)
// All GEMMs run through v_wmma_f32_16x16x32_bf16 with fp32->bf16 LDS staging
// (hardware converts via __builtin_convertvector), vectorized b128 global
// loads, branchless edge handling, and LDS double buffering.
// ---------------------------------------------------------------------------

#define TLEN   2048
#define DMODEL 1024
#define NHEAD  8
#define DHEAD  128
#define NBLK   8
#define DHALF  512
#define DFF    4096
#define SLD    2064   // padded leading dim for the scores matrix (16B aligned)

typedef __attribute__((ext_vector_type(16))) __bf16 v16bf;
typedef __attribute__((ext_vector_type(4)))  __bf16 bf4;
typedef __attribute__((ext_vector_type(8)))  float  v8f;
typedef __attribute__((ext_vector_type(4)))  float  v4f;

union BF16x16 { v16bf v; unsigned int u[8]; };

__device__ __forceinline__ bf4 cvt4bf(v4f a) {
  return __builtin_convertvector(a, bf4);
}

// ---------------------------------------------------------------------------
// Stage one 128x32 A tile and one 32x128 B tile into registers.
// BMODE 0: ldbN == 1  (vectorize B along n, transpose-store into LDS)
// BMODE 1: ldbK == 1  (vectorize B along k, straight packed store)
// Fast path (full tile): unpredicated float4 loads. Edge tiles: clamped
// addresses + cndmask zeroing (no exec-mask branches).
// ---------------------------------------------------------------------------
template <int BMODE>
__device__ __forceinline__ void stage_tiles(
    const float* __restrict__ A, long long lda,
    const float* __restrict__ B, long long ldbK, long long ldbN,
    int mBase, int nBase, int kt, int M, int N, int K,
    v4f ar[4], v4f br[4])
{
  const int tid = threadIdx.x;
  const bool kfull = (kt + 32 <= K);

  // ---- A tile: row = (tid>>3)+32i, col4 = (tid&7)*4 ----
  {
    const int col4 = (tid & 7) * 4;
    if ((mBase + 128 <= M) && kfull) {
#pragma unroll
      for (int i = 0; i < 4; ++i) {
        int row = (tid >> 3) + 32 * i;
        ar[i] = *(const v4f*)&A[(long long)(mBase + row) * lda + kt + col4];
      }
      // speculative prefetch of the tile after next
      __builtin_prefetch(
          &A[(long long)(mBase + (tid >> 3)) * lda + kt + col4 + 64], 0, 1);
    } else {
#pragma unroll
      for (int i = 0; i < 4; ++i) {
        int gm = mBase + (tid >> 3) + 32 * i;
        int cm = gm < M ? gm : (M - 1);
        v4f t;
#pragma unroll
        for (int j = 0; j < 4; ++j) {
          int gk = kt + col4 + j;
          int ck = gk < K ? gk : (K - 1);
          float v = A[(long long)cm * lda + ck];
          t[j] = (gm < M && gk < K) ? v : 0.0f;
        }
        ar[i] = t;
      }
    }
  }

  // ---- B tile ----
  if (BMODE == 0) {  // contiguous along n
    const int n4 = (tid & 31) * 4;
    if ((nBase + 128 <= N) && kfull) {
#pragma unroll
      for (int i = 0; i < 4; ++i) {
        int kk = (tid >> 5) + 8 * i;
        br[i] = *(const v4f*)&B[(long long)(kt + kk) * ldbK + nBase + n4];
      }
    } else {
#pragma unroll
      for (int i = 0; i < 4; ++i) {
        int gk = kt + (tid >> 5) + 8 * i;
        int ck = gk < K ? gk : (K - 1);
        v4f t;
#pragma unroll
        for (int j = 0; j < 4; ++j) {
          int gn = nBase + n4 + j;
          int cn = gn < N ? gn : (N - 1);
          float v = B[(long long)ck * ldbK + cn];
          t[j] = (gk < K && gn < N) ? v : 0.0f;
        }
        br[i] = t;
      }
    }
  } else {           // contiguous along k
    const int k4 = (tid & 7) * 4;
    if ((nBase + 128 <= N) && kfull) {
#pragma unroll
      for (int i = 0; i < 4; ++i) {
        int nn = (tid >> 3) + 32 * i;
        br[i] = *(const v4f*)&B[(long long)(nBase + nn) * ldbN + kt + k4];
      }
    } else {
#pragma unroll
      for (int i = 0; i < 4; ++i) {
        int gn = nBase + (tid >> 3) + 32 * i;
        int cn = gn < N ? gn : (N - 1);
        v4f t;
#pragma unroll
        for (int j = 0; j < 4; ++j) {
          int gk = kt + k4 + j;
          int ck = gk < K ? gk : (K - 1);
          float v = B[(long long)cn * ldbN + ck];
          t[j] = (gn < N && gk < K) ? v : 0.0f;
        }
        br[i] = t;
      }
    }
  }
}

template <int BMODE>
__device__ __forceinline__ void store_tiles(
    __bf16 (*__restrict__ As)[32], __bf16 (*__restrict__ Bs)[32],
    const v4f ar[4], const v4f br[4])
{
  const int tid = threadIdx.x;
  {
    const int col4 = (tid & 7) * 4;
#pragma unroll
    for (int i = 0; i < 4; ++i) {
      int row = (tid >> 3) + 32 * i;
      *(bf4*)&As[row][col4] = cvt4bf(ar[i]);   // one b64 store
    }
  }
  if (BMODE == 0) {
    const int n4 = (tid & 31) * 4;
#pragma unroll
    for (int i = 0; i < 4; ++i) {
      int kk = (tid >> 5) + 8 * i;
      bf4 c = cvt4bf(br[i]);
      Bs[n4 + 0][kk] = c[0];
      Bs[n4 + 1][kk] = c[1];
      Bs[n4 + 2][kk] = c[2];
      Bs[n4 + 3][kk] = c[3];
    }
  } else {
    const int k4 = (tid & 7) * 4;
#pragma unroll
    for (int i = 0; i < 4; ++i) {
      int nn = (tid >> 3) + 32 * i;
      *(bf4*)&Bs[nn][k4] = cvt4bf(br[i]);      // one b64 store
    }
  }
}

// ---------------------------------------------------------------------------
// Tiled bf16 WMMA GEMM: C[M,N] = act(A[M,K] @ B[K,N] + bias) (+ residual)
//   A element (m,k) = A[m*lda + k]          (fp32 -> bf16)
//   B element (k,n) = B[k*ldbK + n*ldbN]    (fp32 -> bf16)
// act (runtime, uniform): 0 none, 1 relu, 2 gelu(tanh).
// Block 256 threads (8 waves), tile 128x128, BK=32, LDS double-buffered;
// wave (wm,wn) owns 64x32 = 4x2 WMMA tiles.
// ---------------------------------------------------------------------------
template <int BMODE>
__global__ __launch_bounds__(256) void gemm_bf16_wmma(
    const float* __restrict__ A, long long lda,
    const float* __restrict__ B, long long ldbK, long long ldbN,
    const float* __restrict__ bias,
    const float* __restrict__ res, long long ldres,
    float* __restrict__ C, long long ldc,
    int M, int N, int K, int act)
{
  __shared__ __bf16 As[2][128][32];   // [buf][m][k]
  __shared__ __bf16 Bs[2][128][32];   // [buf][n][k]

  const int tid  = threadIdx.x;
  const int lane = tid & 31;
  const int wave = tid >> 5;
  const int wm   = wave >> 2;     // 0..1
  const int wn   = wave & 3;      // 0..3
  const int g    = lane >> 4;     // lane group
  const int l15  = lane & 15;
  const int mBase = blockIdx.y * 128;
  const int nBase = blockIdx.x * 128;

  v8f acc[4][2];
#pragma unroll
  for (int mi = 0; mi < 4; ++mi)
#pragma unroll
    for (int ni = 0; ni < 2; ++ni)
#pragma unroll
      for (int e = 0; e < 8; ++e) acc[mi][ni][e] = 0.0f;

  v4f ar[4], br[4];
  stage_tiles<BMODE>(A, lda, B, ldbK, ldbN, mBase, nBase, 0, M, N, K, ar, br);
  store_tiles<BMODE>(As[0], Bs[0], ar, br);
  __syncthreads();

  int cur = 0;
  for (int kt = 0; kt < K; kt += 32) {
    const bool hasNext = (kt + 32) < K;
    if (hasNext)
      stage_tiles<BMODE>(A, lda, B, ldbK, ldbN, mBase, nBase, kt + 32,
                         M, N, K, ar, br);

    // ---- build WMMA fragments (CDNA5 VGPR layout) ----
    BF16x16 af[4], bfr[2];
#pragma unroll
    for (int mi = 0; mi < 4; ++mi) {
      int row = wm * 64 + mi * 16 + l15;
#pragma unroll
      for (int j = 0; j < 4; ++j)
        af[mi].u[j] = *(const unsigned int*)&As[cur][row][g * 8 + 2 * j];
#pragma unroll
      for (int j = 4; j < 8; ++j)
        af[mi].u[j] =
            *(const unsigned int*)&As[cur][row][16 + g * 8 + 2 * (j - 4)];
    }
#pragma unroll
    for (int ni = 0; ni < 2; ++ni) {
      int col = wn * 32 + ni * 16 + l15;
#pragma unroll
      for (int j = 0; j < 4; ++j)
        bfr[ni].u[j] = *(const unsigned int*)&Bs[cur][col][g * 8 + 2 * j];
#pragma unroll
      for (int j = 4; j < 8; ++j)
        bfr[ni].u[j] =
            *(const unsigned int*)&Bs[cur][col][16 + g * 8 + 2 * (j - 4)];
    }

    // ---- 8 WMMAs per K step ----
#pragma unroll
    for (int mi = 0; mi < 4; ++mi)
#pragma unroll
      for (int ni = 0; ni < 2; ++ni)
        acc[mi][ni] = __builtin_amdgcn_wmma_f32_16x16x32_bf16(
            false, af[mi].v, false, bfr[ni].v, (short)0, acc[mi][ni],
            false, false);

    if (hasNext) {
      store_tiles<BMODE>(As[cur ^ 1], Bs[cur ^ 1], ar, br);
      __syncthreads();
      cur ^= 1;
    }
  }

  // ---- epilogue: bias + activation + residual ----
#pragma unroll
  for (int ni = 0; ni < 2; ++ni) {
    int n = nBase + wn * 32 + ni * 16 + l15;
    bool nok = n < N;
    float bv = 0.0f;
    if (bias != nullptr && nok) bv = bias[n];
#pragma unroll
    for (int mi = 0; mi < 4; ++mi) {
#pragma unroll
      for (int e = 0; e < 8; ++e) {
        int m = mBase + wm * 64 + mi * 16 + e + g * 8;
        if (nok && m < M) {
          float v = acc[mi][ni][e] + bv;
          if (act == 1) {
            v = fmaxf(v, 0.0f);
          } else if (act == 2) {
            float x3 = v * v * v;
            v = 0.5f * v * (1.0f + tanhf(0.7978845608028654f *
                                         (v + 0.044715f * x3)));
          }
          if (res) v += res[(long long)m * ldres + n];
          C[(long long)m * ldc + n] = v;
        }
      }
    }
  }
}

// ---------------------------------------------------------------------------
// LayerNorm over D=1024 (one block per row)
// ---------------------------------------------------------------------------
__global__ __launch_bounds__(256) void layernorm_kernel(
    const float* __restrict__ x, const float* __restrict__ gamma,
    const float* __restrict__ beta, float* __restrict__ out, float eps)
{
  const int row = blockIdx.x;
  const float* xr = x + (long long)row * DMODEL;
  float s = 0.0f, sq = 0.0f;
  for (int i = threadIdx.x; i < DMODEL; i += 256) {
    float v = xr[i];
    s += v; sq += v * v;
  }
  __shared__ float r1[256], r2[256];
  r1[threadIdx.x] = s; r2[threadIdx.x] = sq;
  __syncthreads();
  for (int st = 128; st > 0; st >>= 1) {
    if (threadIdx.x < st) {
      r1[threadIdx.x] += r1[threadIdx.x + st];
      r2[threadIdx.x] += r2[threadIdx.x + st];
    }
    __syncthreads();
  }
  float mean = r1[0] * (1.0f / DMODEL);
  float var  = r2[0] * (1.0f / DMODEL) - mean * mean;
  float inv  = rsqrtf(var + eps);
  for (int i = threadIdx.x; i < DMODEL; i += 256)
    out[(long long)row * DMODEL + i] = (xr[i] - mean) * inv * gamma[i] + beta[i];
}

// ---------------------------------------------------------------------------
// Row softmax with scale (in-place), one block per row.
// ---------------------------------------------------------------------------
__global__ __launch_bounds__(256) void softmax_rows(
    float* __restrict__ S, int N, long long ld, float scale)
{
  float* r = S + (long long)blockIdx.x * ld;
  __shared__ float red[256];
  float mx = -3.0e38f;
  for (int i = threadIdx.x; i < N; i += 256) mx = fmaxf(mx, r[i] * scale);
  red[threadIdx.x] = mx;
  __syncthreads();
  for (int st = 128; st > 0; st >>= 1) {
    if (threadIdx.x < st)
      red[threadIdx.x] = fmaxf(red[threadIdx.x], red[threadIdx.x + st]);
    __syncthreads();
  }
  mx = red[0];
  __syncthreads();
  float s = 0.0f;
  for (int i = threadIdx.x; i < N; i += 256) {
    float e = __expf(r[i] * scale - mx);
    r[i] = e;
    s += e;
  }
  red[threadIdx.x] = s;
  __syncthreads();
  for (int st = 128; st > 0; st >>= 1) {
    if (threadIdx.x < st) red[threadIdx.x] += red[threadIdx.x + st];
    __syncthreads();
  }
  float inv = 1.0f / red[0];
  for (int i = threadIdx.x; i < N; i += 256) r[i] *= inv;
}

// write the learned bias_k / bias_v token as row T of K/V
__global__ void write_bias_row(float* __restrict__ kbuf, float* __restrict__ vbuf,
                               const float* __restrict__ bk,
                               const float* __restrict__ bv)
{
  int i = blockIdx.x * 256 + threadIdx.x;
  if (i < DMODEL) {
    kbuf[(long long)TLEN * DMODEL + i] = bk[i];
    vbuf[(long long)TLEN * DMODEL + i] = bv[i];
  }
}

// locate head: dot(row[512], W2[512]) + b2 -> sigmoid
__global__ __launch_bounds__(256) void locate_dot(
    const float* __restrict__ hid, const float* __restrict__ w2,
    const float* __restrict__ b2, float* __restrict__ located)
{
  const int t = blockIdx.x;
  float s = 0.0f;
  for (int i = threadIdx.x; i < DHALF; i += 256)
    s += hid[(long long)t * DHALF + i] * w2[i];
  __shared__ float red[256];
  red[threadIdx.x] = s;
  __syncthreads();
  for (int st = 128; st > 0; st >>= 1) {
    if (threadIdx.x < st) red[threadIdx.x] += red[threadIdx.x + st];
    __syncthreads();
  }
  if (threadIdx.x == 0)
    located[t] = 1.0f / (1.0f + __expf(-(red[0] + b2[0])));
}

__global__ void zero_kernel(float* __restrict__ p, long long n)
{
  long long i = (long long)blockIdx.x * 256 + threadIdx.x;
  if (i < n) p[i] = 0.0f;
}

// run-length segmentation (serial scan; T=2048 is tiny)
__global__ void seg_scan(const float* __restrict__ located,
                         const float* __restrict__ thr,
                         int* __restrict__ clips, int* __restrict__ segids,
                         int* __restrict__ numseg)
{
  if (blockIdx.x == 0 && threadIdx.x == 0) {
    float th = thr[0];
    int sid = 0, prev = 0;
    for (int t = 0; t < TLEN; ++t) {
      int c = located[t] > th ? 1 : 0;
      if (t > 0 && c != prev) sid++;
      clips[t] = c;
      segids[t] = sid;
      prev = c;
    }
    numseg[0] = sid + 1;
  }
}

__global__ void count_accum(const int* __restrict__ segids,
                            const int* __restrict__ clips,
                            float* __restrict__ counts,
                            int* __restrict__ segtypes)
{
  int t = blockIdx.x * 256 + threadIdx.x;
  if (t < TLEN) {
    int s = segids[t];
    atomicAdd(&counts[s], 1.0f);
    atomicMax(&segtypes[s], clips[t]);
  }
}

__global__ void pool_accum(const int* __restrict__ segids,
                           const float* __restrict__ vAct,
                           const float* __restrict__ aAct,
                           float* __restrict__ vid_sum,
                           float* __restrict__ aud_sum)
{
  long long idx = (long long)blockIdx.x * 256 + threadIdx.x;
  const long long TD = (long long)TLEN * DMODEL;
  if (idx < TD) {
    int t = (int)(idx / DMODEL);
    int d = (int)(idx % DMODEL);
    int s = segids[t];
    atomicAdd(&vid_sum[(long long)s * DMODEL + d], vAct[idx]);
    atomicAdd(&aud_sum[(long long)s * DMODEL + d], aAct[idx]);
  }
}

__global__ void pool_final(const float* __restrict__ vid_sum,
                           const float* __restrict__ aud_sum,
                           const float* __restrict__ counts,
                           float* __restrict__ out)
{
  long long idx = (long long)blockIdx.x * 256 + threadIdx.x;
  const long long TD = (long long)TLEN * DMODEL;
  if (idx < TD) {
    int s = (int)(idx / DMODEL);
    float den = fmaxf(counts[s], 1.0f);
    out[idx]      = vid_sum[idx] / den;
    out[TD + idx] = aud_sum[idx] / den;
  }
}

__global__ void seg_out_tail(const float* __restrict__ located,
                             const int* __restrict__ segids,
                             const int* __restrict__ segtypes,
                             const int* __restrict__ numseg,
                             float* __restrict__ out)
{
  int t = blockIdx.x * 256 + threadIdx.x;
  const long long TD2 = 2LL * TLEN * DMODEL;
  if (t < TLEN) {
    out[TD2 + t]            = located[t];
    out[TD2 + TLEN + t]     = (float)segids[t];
    out[TD2 + 2 * TLEN + t] = (float)segtypes[t];
  }
  if (t == 0) out[TD2 + 3 * TLEN] = (float)numseg[0];
}

// ---------------------------------------------------------------------------
// host-side dispatch
// ---------------------------------------------------------------------------
static inline void launch_gemm(hipStream_t st, const float* A, long long lda,
                               const float* B, long long ldbK, long long ldbN,
                               const float* bias, const float* res,
                               long long ldres, float* C, long long ldc,
                               int M, int N, int K, int act, int bmode)
{
  dim3 grid((unsigned)((N + 127) / 128), (unsigned)((M + 127) / 128));
  dim3 block(256);
  if (bmode == 1)
    gemm_bf16_wmma<1><<<grid, block, 0, st>>>(A, lda, B, ldbK, ldbN, bias,
                                              res, ldres, C, ldc, M, N, K, act);
  else
    gemm_bf16_wmma<0><<<grid, block, 0, st>>>(A, lda, B, ldbK, ldbN, bias,
                                              res, ldres, C, ldc, M, N, K, act);
}

extern "C" void kernel_launch(void* const* d_in, const int* in_sizes, int n_in,
                              void* d_out, int out_size, void* d_ws,
                              size_t ws_size, hipStream_t stream)
{
  (void)in_sizes; (void)n_in; (void)out_size; (void)ws_size;

  const float* video   = (const float*)d_in[0];   // [T,1024]
  const float* audio   = (const float*)d_in[1];   // [T,128]
  const float* ap_W1   = (const float*)d_in[2];
  const float* ap_b1   = (const float*)d_in[3];
  const float* ap_W2   = (const float*)d_in[4];
  const float* ap_b2   = (const float*)d_in[5];
  const float* vp_W1   = (const float*)d_in[6];
  const float* vp_b1   = (const float*)d_in[7];
  const float* vp_W2   = (const float*)d_in[8];
  const float* vp_b2   = (const float*)d_in[9];
  const float* pre_g   = (const float*)d_in[10];
  const float* pre_b   = (const float*)d_in[11];
  const float* blk_Wq  = (const float*)d_in[12];
  const float* blk_bq  = (const float*)d_in[13];
  const float* blk_Wk  = (const float*)d_in[14];
  const float* blk_bk  = (const float*)d_in[15];
  const float* blk_Wv  = (const float*)d_in[16];
  const float* blk_bv  = (const float*)d_in[17];
  const float* blk_Wo  = (const float*)d_in[18];
  const float* blk_bo  = (const float*)d_in[19];
  const float* blk_bkx = (const float*)d_in[20];
  const float* blk_bvx = (const float*)d_in[21];
  const float* blk_g1  = (const float*)d_in[22];
  const float* blk_b1  = (const float*)d_in[23];
  const float* blk_g2  = (const float*)d_in[24];
  const float* blk_b2  = (const float*)d_in[25];
  const float* blk_mW1 = (const float*)d_in[26];
  const float* blk_mb1 = (const float*)d_in[27];
  const float* blk_mW2 = (const float*)d_in[28];
  const float* blk_mb2 = (const float*)d_in[29];
  const float* lh_W1   = (const float*)d_in[30];
  const float* lh_b1   = (const float*)d_in[31];
  const float* lh_W2   = (const float*)d_in[32];
  const float* lh_b2   = (const float*)d_in[33];
  const float* thr     = (const float*)d_in[34];

  float* out = (float*)d_out;

  // workspace layout (floats)
  const long long TD = (long long)TLEN * DMODEL;
  float* w = (float*)d_ws;
  size_t off = 0;
  float* aAct    = w + off; off += (size_t)TD;
  float* vAct    = w + off; off += (size_t)TD;
  float* x       = w + off; off += (size_t)TD;
  float* h       = w + off; off += (size_t)TD;
  float* q       = w + off; off += (size_t)TD;
  float* kbuf    = w + off; off += (size_t)(TLEN + 1) * DMODEL;
  float* vbuf    = w + off; off += (size_t)(TLEN + 1) * DMODEL;
  float* attnO   = w + off; off += (size_t)TD;
  float* scores  = w + off; off += (size_t)TLEN * SLD;
  float* mlpH    = w + off; off += (size_t)TLEN * DFF;
  float* tmpH    = w + off; off += (size_t)TLEN * DHALF;
  float* vid_sum = w + off; off += (size_t)TD;
  float* aud_sum = w + off; off += (size_t)TD;
  float* counts  = w + off; off += (size_t)TLEN;
  int*   segtypes= (int*)(w + off); off += (size_t)TLEN;
  int*   clips   = (int*)(w + off); off += (size_t)TLEN;
  int*   segids  = (int*)(w + off); off += (size_t)TLEN;
  int*   numseg  = (int*)(w + off); off += 1;
  float* located = w + off; off += (size_t)TLEN;

  const float attn_scale = 0.08838834764831845f;  // 1/sqrt(128)

  // ---- projectors: Linear -> ReLU -> Linear ----
  launch_gemm(stream, audio, 128, ap_W1, DHALF, 1, ap_b1, nullptr, 0,
              tmpH, DHALF, TLEN, DHALF, 128, 1, 0);
  launch_gemm(stream, tmpH, DHALF, ap_W2, DMODEL, 1, ap_b2, nullptr, 0,
              aAct, DMODEL, TLEN, DMODEL, DHALF, 0, 0);
  launch_gemm(stream, video, DMODEL, vp_W1, DHALF, 1, vp_b1, nullptr, 0,
              tmpH, DHALF, TLEN, DHALF, DMODEL, 1, 0);
  launch_gemm(stream, tmpH, DHALF, vp_W2, DMODEL, 1, vp_b2, nullptr, 0,
              vAct, DMODEL, TLEN, DMODEL, DHALF, 0, 0);

  // pre-transformer LayerNorm (eps 1e-6): x = ln(a)
  layernorm_kernel<<<TLEN, 256, 0, stream>>>(aAct, pre_g, pre_b, x, 1e-6f);

  // ---- transformer blocks ----
  for (int l = 0; l < NBLK; ++l) {
    const long long WOFF = (long long)l * DMODEL * DMODEL;
    const long long BOFF = (long long)l * DMODEL;
    // ln1
    layernorm_kernel<<<TLEN, 256, 0, stream>>>(x, blk_g1 + BOFF, blk_b1 + BOFF,
                                               h, 1e-5f);
    // q = h @ Wq + bq
    launch_gemm(stream, h, DMODEL, blk_Wq + WOFF, DMODEL, 1, blk_bq + BOFF,
                nullptr, 0, q, DMODEL, TLEN, DMODEL, DMODEL, 0, 0);
    // k/v from projected video
    launch_gemm(stream, vAct, DMODEL, blk_Wk + WOFF, DMODEL, 1, blk_bk + BOFF,
                nullptr, 0, kbuf, DMODEL, TLEN, DMODEL, DMODEL, 0, 0);
    launch_gemm(stream, vAct, DMODEL, blk_Wv + WOFF, DMODEL, 1, blk_bv + BOFF,
                nullptr, 0, vbuf, DMODEL, TLEN, DMODEL, DMODEL, 0, 0);
    // add_bias_kv token (row T)
    write_bias_row<<<(DMODEL + 255) / 256, 256, 0, stream>>>(
        kbuf, vbuf, blk_bkx + BOFF, blk_bvx + BOFF);

    // attention per head
    for (int hh = 0; hh < NHEAD; ++hh) {
      const float* Qh = q + hh * DHEAD;
      const float* Kh = kbuf + hh * DHEAD;
      const float* Vh = vbuf + hh * DHEAD;
      // scores = Q_h [T,DH] @ K_h^T [DH,T+1] ; B(k=d, n=s) = K[s*D + d]
      launch_gemm(stream, Qh, DMODEL, Kh, 1, DMODEL, nullptr, nullptr, 0,
                  scores, SLD, TLEN, TLEN + 1, DHEAD, 0, /*BMODE=*/1);
      softmax_rows<<<TLEN, 256, 0, stream>>>(scores, TLEN + 1, SLD,
                                             attn_scale);
      // O_h = P [T,T+1] @ V_h [T+1,DH] ; B(k=s, n=d) = V[s*D + d]
      launch_gemm(stream, scores, SLD, Vh, DMODEL, 1, nullptr, nullptr, 0,
                  attnO + hh * DHEAD, DMODEL, TLEN, DHEAD, TLEN + 1, 0, 0);
    }

    // x = x + attnO @ Wo + bo
    launch_gemm(stream, attnO, DMODEL, blk_Wo + WOFF, DMODEL, 1,
                blk_bo + BOFF, x, DMODEL, x, DMODEL, TLEN, DMODEL, DMODEL,
                0, 0);
    // ln2
    layernorm_kernel<<<TLEN, 256, 0, stream>>>(x, blk_g2 + BOFF, blk_b2 + BOFF,
                                               h, 1e-5f);
    // mlpH = gelu(h @ mW1 + mb1)
    launch_gemm(stream, h, DMODEL, blk_mW1 + (long long)l * DMODEL * DFF, DFF,
                1, blk_mb1 + (long long)l * DFF, nullptr, 0, mlpH, DFF, TLEN,
                DFF, DMODEL, 2, 0);
    // x = x + mlpH @ mW2 + mb2
    launch_gemm(stream, mlpH, DFF, blk_mW2 + (long long)l * DFF * DMODEL,
                DMODEL, 1, blk_mb2 + BOFF, x, DMODEL, x, DMODEL, TLEN, DMODEL,
                DFF, 0, 0);
  }

  // ---- locate head ----
  launch_gemm(stream, x, DMODEL, lh_W1, DHALF, 1, lh_b1, nullptr, 0, tmpH,
              DHALF, TLEN, DHALF, DMODEL, 1, 0);
  locate_dot<<<TLEN, 256, 0, stream>>>(tmpH, lh_W2, lh_b2, located);

  // ---- segmentation + pooling ----
  const long long zn = 2 * TD + 2 * TLEN;  // vid_sum, aud_sum, counts, segtypes
  zero_kernel<<<(unsigned)((zn + 255) / 256), 256, 0, stream>>>(vid_sum, zn);
  seg_scan<<<1, 32, 0, stream>>>(located, thr, clips, segids, numseg);
  count_accum<<<(TLEN + 255) / 256, 256, 0, stream>>>(segids, clips, counts,
                                                      segtypes);
  pool_accum<<<(unsigned)((TD + 255) / 256), 256, 0, stream>>>(
      segids, vAct, aAct, vid_sum, aud_sum);
  pool_final<<<(unsigned)((TD + 255) / 256), 256, 0, stream>>>(
      vid_sum, aud_sum, counts, out);
  seg_out_tail<<<(TLEN + 255) / 256, 256, 0, stream>>>(located, segids,
                                                       segtypes, numseg, out);
}
